// GNNModel_with_leaky_ReLU_and_Attention_29154238005711
// MI455X (gfx1250) — compile-verified, matrix operand order checked by reading, not run on hardware
//
#include <hip/hip_runtime.h>
#include <hip/hip_bf16.h>

// ---------------- problem constants (match reference) ----------------
#define NSITES 100000
#define NWELLS 400000
#define NEDGES 500000
#define DIN    64
#define HEADS  4

typedef __attribute__((ext_vector_type(16))) __bf16 v16bf;
typedef __attribute__((ext_vector_type(8)))  float  v8f;

#define TRY_ASYNC_LDS 1

// ---------------- helpers ----------------
__device__ __forceinline__ unsigned enc_ord(float f) {
  unsigned u = __float_as_uint(f);
  return (u & 0x80000000u) ? ~u : (u | 0x80000000u);
}
__device__ __forceinline__ float dec_ord(unsigned u) {
  return (u & 0x80000000u) ? __uint_as_float(u & 0x7FFFFFFFu)
                           : __uint_as_float(~u);
}
#define ENC_NEG_INF 0x007FFFFFu  // enc_ord(-inf)

// =====================================================================
// Fragment packing.
// WMMA bf16 A-fragment layout (wave32, 16x32 tile): lane l, half=l>>4, m=l&15:
//   j in [0,8)  -> K = k0 + half*8 + j
//   j in [8,16) -> K = k0 + 16 + half*8 + (j-8)
// We pre-pack A[M,64] (f32, row-major) into Apk: per (tileM, kk, lane) a
// contiguous 16-bf16 (32B) chunk -> GEMM A load = 2x global_load_b128.
// =====================================================================
__global__ void pack_A(const float* __restrict__ X, __bf16* __restrict__ Apk, int M) {
  int t = blockIdx.x * blockDim.x + threadIdx.x;  // one per (row, K-octet)
  if (t >= M * 8) return;
  int n = t >> 3, oct = t & 7;
  int tileM = n >> 4, m = n & 15;
  int kk = oct >> 2, o2 = oct & 3;
  int half = o2 & 1, jbase = (o2 >> 1) * 8;
  int lane = m + 16 * half;
  const float* src = X + (size_t)n * DIN + oct * 8;
  union { __bf16 h[8]; uint4 q; } tmp;
#pragma unroll
  for (int j = 0; j < 8; ++j) tmp.h[j] = (__bf16)src[j];
  *(uint4*)(Apk + (((size_t)(tileM * 2 + kk) * 32 + lane) * 16 + jbase)) = tmp.q;
}

// B[64,N] f32 row-major -> Bpk fragments: index ((tileN*2+kk)*32+lane)*16+j,
// lane holds column n = lane&15, same K striping as A.
__global__ void pack_B(const float* __restrict__ B, __bf16* __restrict__ Bpk, int N) {
  int t = blockIdx.x * blockDim.x + threadIdx.x;
  if (t >= (N / 16) * 1024) return;
  int j = t & 15, lane = (t >> 4) & 31, kk = (t >> 9) & 1, tileN = t >> 10;
  int half = lane >> 4, m16 = lane & 15;
  int K = kk * 32 + half * 8 + (j < 8 ? j : 16 + (j - 8));
  Bpk[t] = (__bf16)B[(size_t)K * N + tileN * 16 + m16];
}

// =====================================================================
// WMMA GEMM: Out[M,N] = Apk(M,64) @ Bpk(64,N) (+Cin)(+bias), f32 accumulate.
// N is compile-time -> row strides become immediate store offsets.
// 8 waves/block along M share one tileN -> B panel staged once through LDS
// (async global->LDS when the builtin exists).
// =====================================================================
template <int N, bool HASC, bool HASBIAS>
__global__ __launch_bounds__(256)
void wmma_gemm_k64(const __bf16* __restrict__ Apk, const __bf16* __restrict__ Bpk,
                   const float* __restrict__ Cin, const float* __restrict__ bias,
                   float* __restrict__ Out, int M) {
  __shared__ uint4 sB[128];  // 2 k-steps * 32 lanes * 32B = 2KB
  const int tileN = blockIdx.y;
  const int t = threadIdx.y * 32 + threadIdx.x;
  const uint4* bsrc = (const uint4*)(Bpk + (size_t)tileN * 1024);
#if TRY_ASYNC_LDS && defined(__HIP_DEVICE_COMPILE__) && \
    __has_builtin(__builtin_amdgcn_global_load_async_to_lds_b128) && \
    __has_builtin(__builtin_amdgcn_s_wait_asynccnt)
  typedef int v4i_ld __attribute__((__vector_size__(4 * sizeof(int))));
  typedef __attribute__((address_space(1))) v4i_ld gvec_t;
  typedef __attribute__((address_space(3))) v4i_ld svec_t;
  if (t < 128)
    __builtin_amdgcn_global_load_async_to_lds_b128(
        (gvec_t*)(bsrc + t), (svec_t*)(&sB[t]), 0, 0);
  __builtin_amdgcn_s_wait_asynccnt(0);
  __syncthreads();
#else
  if (t < 128) sB[t] = bsrc[t];
  __syncthreads();
#endif

  const int tileM = blockIdx.x * 8 + threadIdx.y;
  if (tileM * 16 >= M) return;  // uniform per-wave exit (after barrier)
  const int lane = threadIdx.x;
  const int half = lane >> 4;
  const int m16  = lane & 15;
  const int colBase = tileN * 16;

  v8f c = {};
  if (HASC) {
    const float* cbase = Cin + (size_t)(tileM * 16 + 8 * half) * N + colBase + m16;
#pragma unroll
    for (int r = 0; r < 8; ++r) c[r] = cbase[r * N];  // constant offsets
  }

#pragma unroll
  for (int kk = 0; kk < 2; ++kk) {
    union { uint4 q[2]; v16bf v; } ua, ub;
    const uint4* ap = (const uint4*)(Apk + ((size_t)(tileM * 2 + kk) * 32 + lane) * 16);
    ua.q[0] = ap[0];
    ua.q[1] = ap[1];
    ub.q[0] = sB[(kk * 32 + lane) * 2 + 0];
    ub.q[1] = sB[(kk * 32 + lane) * 2 + 1];
    c = __builtin_amdgcn_wmma_f32_16x16x32_bf16(false, ua.v, false, ub.v,
                                                (short)0, c, false, false);
  }

  float bv = 0.f;
  if (HASBIAS) bv = bias[colBase + m16];
  float* obase = Out + (size_t)(tileM * 16 + 8 * half) * N + colBase + m16;
#pragma unroll
  for (int r = 0; r < 8; ++r) {
    float v = c[r];
    if (HASBIAS) v += bv;
    obase[r * N] = v;  // constant offsets
  }
}

// ---------------- small-weight precomputation ----------------
__global__ void prep_weights(const float* __restrict__ W_sw,
                             const float* __restrict__ att_s,
                             const float* __restrict__ att_d,
                             const float* __restrict__ Wl_ww,
                             const float* __restrict__ Wr_ww,
                             const float* __restrict__ Wr_ws,
                             const float* __restrict__ W_ss,
                             const float* __restrict__ b_sw,
                             const float* __restrict__ bl_ww,
                             const float* __restrict__ bl_ws,
                             const float* __restrict__ b_ss,
                             float* __restrict__ Wc_ww, float* __restrict__ Wc_site,
                             float* __restrict__ w_as, float* __restrict__ w_ad,
                             float* __restrict__ bias_well, float* __restrict__ bias_site) {
  int t = blockIdx.x * blockDim.x + threadIdx.x;
  if (t < 4096) {
    int k = t >> 6, cc = t & 63;
    Wc_ww[t] = Wl_ww[t] + Wr_ww[t];
    float s4 = 0.f;
#pragma unroll
    for (int h = 0; h < HEADS; ++h) s4 += W_ss[k * 256 + h * 64 + cc];
    Wc_site[t] = Wr_ws[t] + 0.25f * s4;
  }
  if (t < 256) {
    int k = t >> 2, h = t & 3;
    float as = 0.f, ad = 0.f;
    for (int cc = 0; cc < 64; ++cc) {
      float w = W_sw[k * 256 + h * 64 + cc];
      as += w * att_s[h * 64 + cc];
      ad += w * att_d[h * 64 + cc];
    }
    w_as[t] = as;
    w_ad[t] = ad;
  }
  if (t < 64) {
    bias_well[t] = bl_ww[t] + b_sw[t];
    bias_site[t] = bl_ws[t] + b_ss[t];
  }
}

// ---------------- generic fill ----------------
__global__ void fill_u32(unsigned* __restrict__ p, unsigned val, long long n) {
  long long stride = (long long)gridDim.x * blockDim.x;
  for (long long i = blockIdx.x * (long long)blockDim.x + threadIdx.x; i < n; i += stride)
    p[i] = val;
}

// ---------------- attention logits: al[n,h] = x[n,:] @ Wsm[:,h] ----------------
__global__ void matvec4(const float* __restrict__ X, const float* __restrict__ Wsm,
                        float* __restrict__ al, int Nn) {
  int n = blockIdx.x * blockDim.x + threadIdx.x;
  if (n >= Nn) return;
  const float* xr = X + (size_t)n * DIN;
  float acc0 = 0.f, acc1 = 0.f, acc2 = 0.f, acc3 = 0.f;
#pragma unroll
  for (int k = 0; k < DIN; ++k) {
    float xv = xr[k];
    acc0 += xv * Wsm[k * 4 + 0];
    acc1 += xv * Wsm[k * 4 + 1];
    acc2 += xv * Wsm[k * 4 + 2];
    acc3 += xv * Wsm[k * 4 + 3];
  }
  float* o = al + (size_t)n * 4;
  o[0] = acc0; o[1] = acc1; o[2] = acc2; o[3] = acc3;
}

// ---------------- GAT pass 1: segment max (ordered-uint atomic max) ----------------
__global__ void gat_edge_max(const int* __restrict__ src, const int* __restrict__ dst,
                             const float* __restrict__ al_s, const float* __restrict__ al_d,
                             unsigned* __restrict__ mmax, int E_) {
  int e = blockIdx.x * blockDim.x + threadIdx.x;
  if (e >= E_) return;
  int s = src[e], d = dst[e];
#pragma unroll
  for (int h = 0; h < HEADS; ++h) {
    float v = al_s[(size_t)s * 4 + h] + al_d[(size_t)d * 4 + h];
    v = v > 0.f ? v : 0.2f * v;  // leaky_relu(0.2)
    atomicMax(&mmax[(size_t)d * 4 + h], enc_ord(v));
  }
}

// ---------------- GAT pass 2: exp(e - m), denominator ----------------
__global__ void gat_edge_exp(const int* __restrict__ src, const int* __restrict__ dst,
                             const float* __restrict__ al_s, const float* __restrict__ al_d,
                             const unsigned* __restrict__ mmax,
                             float* __restrict__ exbuf, float* __restrict__ denom, int E_) {
  int e = blockIdx.x * blockDim.x + threadIdx.x;
  if (e >= E_) return;
  int s = src[e], d = dst[e];
#pragma unroll
  for (int h = 0; h < HEADS; ++h) {
    float v = al_s[(size_t)s * 4 + h] + al_d[(size_t)d * 4 + h];
    v = v > 0.f ? v : 0.2f * v;
    float ex = __expf(v - dec_ord(mmax[(size_t)d * 4 + h]));
    exbuf[(size_t)e * 4 + h] = ex;
    atomicAdd(&denom[(size_t)d * 4 + h], ex);
  }
}

// denom -> 0.25/denom (wells with no edges produce inf, never read)
__global__ void recip_scale(float* __restrict__ denom, long long n) {
  long long i = blockIdx.x * (long long)blockDim.x + threadIdx.x;
  if (i >= n) return;
  denom[i] = 0.25f / denom[i];
}

// ---------------- GAT pass 3: scatter messages, fused head-mean ----------------
// acc_w[d,c] += sum_h ex_h * rden_h * xs[s, h*64+c]  (rden = 0.25/denom)
__global__ void gat_edge_scatter(const int* __restrict__ src, const int* __restrict__ dst,
                                 const float* __restrict__ xs,
                                 const float* __restrict__ exbuf,
                                 const float* __restrict__ rden,
                                 float* __restrict__ acc_w, long long total) {
  long long gid = blockIdx.x * (long long)blockDim.x + threadIdx.x;
  if (gid >= total) return;
  int e = (int)(gid >> 6);
  int cc = (int)(gid & 63);
  int s = src[e], d = dst[e];
  const float* xrow = xs + (size_t)s * 256;
  __builtin_prefetch(xrow + cc, 0, 0);  // global_prefetch_b8
  float msum = 0.f;
#pragma unroll
  for (int h = 0; h < HEADS; ++h) {
    float alpha = exbuf[(size_t)e * 4 + h] * rden[(size_t)d * 4 + h];
    msum += alpha * xrow[h * 64 + cc];
  }
  atomicAdd(&acc_w[(size_t)d * 64 + cc], msum);
}

// ---------------- SAGE edge scatter (wells -> sites) ----------------
__global__ void sage_edge(const int* __restrict__ src, const int* __restrict__ dst,
                          const float* __restrict__ xw,
                          float* __restrict__ ssum, float* __restrict__ cnt, long long total) {
  long long gid = blockIdx.x * (long long)blockDim.x + threadIdx.x;
  if (gid >= total) return;
  int e = (int)(gid >> 6);
  int cc = (int)(gid & 63);
  int s = src[e], d = dst[e];
  __builtin_prefetch(xw + (size_t)s * 64 + cc, 0, 0);
  atomicAdd(&ssum[(size_t)d * 64 + cc], xw[(size_t)s * 64 + cc]);
  if (cc == 0) atomicAdd(&cnt[d], 1.0f);
}

// ---------------- SAGE mean (in place) ----------------
__global__ void sage_mean(float* __restrict__ ssum, const float* __restrict__ cnt, long long total) {
  long long gid = blockIdx.x * (long long)blockDim.x + threadIdx.x;
  if (gid >= total) return;
  int n = (int)(gid >> 6);
  ssum[gid] *= (1.0f / fmaxf(cnt[n], 1.0f));
}

// ---------------- output heads: relu -> dot -> leaky_relu(0.01) ----------------
__global__ void head_kernel(const float* __restrict__ feat, const float* __restrict__ Wv,
                            const float* __restrict__ bv, float* __restrict__ out, int Nn) {
  int n = blockIdx.x * blockDim.x + threadIdx.x;
  if (n >= Nn) return;
  const float* fr = feat + (size_t)n * 64;
  float o = bv[0];
#pragma unroll
  for (int cc = 0; cc < 64; ++cc) {
    float v = fmaxf(fr[cc], 0.f);
    o += v * Wv[cc];
  }
  out[n] = o > 0.f ? o : 0.01f * o;
}

// ---------------- launcher ----------------
extern "C" void kernel_launch(void* const* d_in, const int* in_sizes, int n_in,
                              void* d_out, int out_size, void* d_ws, size_t ws_size,
                              hipStream_t stream) {
  const float* x_sites  = (const float*)d_in[0];
  const float* x_wells  = (const float*)d_in[1];
  const int*   e_sw_src = (const int*)d_in[2];
  const int*   e_sw_dst = (const int*)d_in[3];
  const int*   e_ws_src = (const int*)d_in[4];
  const int*   e_ws_dst = (const int*)d_in[5];
  const float* W_sw     = (const float*)d_in[6];
  const float* att_s_sw = (const float*)d_in[7];
  const float* att_d_sw = (const float*)d_in[8];
  const float* b_sw     = (const float*)d_in[9];
  const float* Wl_ws    = (const float*)d_in[10];
  const float* bl_ws    = (const float*)d_in[11];
  const float* Wr_ws    = (const float*)d_in[12];
  const float* W_ss     = (const float*)d_in[13];
  // d_in[14], d_in[15] (att_*_ss) unused: self-loop softmax == 1
  const float* b_ss     = (const float*)d_in[16];
  const float* Wl_ww    = (const float*)d_in[17];
  const float* bl_ww    = (const float*)d_in[18];
  const float* Wr_ww    = (const float*)d_in[19];
  const float* Wg       = (const float*)d_in[20];
  const float* bg       = (const float*)d_in[21];
  const float* Wsit     = (const float*)d_in[22];
  const float* bsit     = (const float*)d_in[23];

  float* o_gw = (float*)d_out;  // [NWELLS]
  float* o_si = o_gw + NWELLS;  // [NSITES]

  // scratch layout (float units)
  float* ws = (float*)d_ws;
  size_t off = 0;
  float*    xs       = ws + off; off += (size_t)NSITES * 256;  // x_sites @ W_sw
  float*    acc_w    = ws + off; off += (size_t)NWELLS * 64;   // well accumulator
  float*    al_s     = ws + off; off += (size_t)NSITES * 4;
  float*    al_d     = ws + off; off += (size_t)NWELLS * 4;
  unsigned* mmax     = (unsigned*)(ws + off); off += (size_t)NWELLS * 4;
  float*    denom    = ws + off; off += (size_t)NWELLS * 4;
  float*    exbuf    = ws + off; off += (size_t)NEDGES * 4;
  float*    ssum     = ws + off; off += (size_t)NSITES * 64;
  float*    cnt      = ws + off; off += (size_t)NSITES;
  float*    site_lin = ws + off; off += (size_t)NSITES * 64;
  float*    Wc_ww    = ws + off; off += 64 * 64;
  float*    Wc_site  = ws + off; off += 64 * 64;
  float*    w_as     = ws + off; off += 64 * 4;
  float*    w_ad     = ws + off; off += 64 * 4;
  float*    biasWell = ws + off; off += 64;
  float*    biasSite = ws + off; off += 64;
  // bf16 fragment buffers (2 bf16 per float slot)
  __bf16* ApkSites = (__bf16*)(ws + off); off += (size_t)NSITES * 32;  // NSITES*64 bf16
  __bf16* ApkWells = (__bf16*)(ws + off); off += (size_t)NWELLS * 32;
  __bf16* ApkMean  = (__bf16*)(ws + off); off += (size_t)NSITES * 32;
  __bf16* BpkSW    = (__bf16*)(ws + off); off += 16 * 512;  // 16 tiles * 1024 bf16
  __bf16* BpkWW    = (__bf16*)(ws + off); off += 4 * 512;
  __bf16* BpkSite  = (__bf16*)(ws + off); off += 4 * 512;
  __bf16* BpkLWS   = (__bf16*)(ws + off); off += 4 * 512;

  // 1. fused small weights, then B-fragment packs
  prep_weights<<<16, 256, 0, stream>>>(W_sw, att_s_sw, att_d_sw, Wl_ww, Wr_ww, Wr_ws, W_ss,
                                       b_sw, bl_ww, bl_ws, b_ss,
                                       Wc_ww, Wc_site, w_as, w_ad, biasWell, biasSite);
  pack_B<<<64, 256, 0, stream>>>(W_sw, BpkSW, 256);
  pack_B<<<16, 256, 0, stream>>>(Wc_ww, BpkWW, 64);
  pack_B<<<16, 256, 0, stream>>>(Wc_site, BpkSite, 64);
  pack_B<<<16, 256, 0, stream>>>(Wl_ws, BpkLWS, 64);

  // 2. A-fragment packs (bf16, WMMA layout)
  pack_A<<<(NSITES * 8 + 255) / 256, 256, 0, stream>>>(x_sites, ApkSites, NSITES);
  pack_A<<<(NWELLS * 8 + 255) / 256, 256, 0, stream>>>(x_wells, ApkWells, NWELLS);

  // 3. init buffers
  fill_u32<<<1024, 256, 0, stream>>>(mmax, ENC_NEG_INF, (long long)NWELLS * 4);
  fill_u32<<<1024, 256, 0, stream>>>((unsigned*)denom, 0u, (long long)NWELLS * 4);
  fill_u32<<<1024, 256, 0, stream>>>((unsigned*)ssum, 0u, (long long)NSITES * 64);
  fill_u32<<<1024, 256, 0, stream>>>((unsigned*)cnt, 0u, (long long)NSITES);

  // 4. attention logits
  matvec4<<<(NSITES + 255) / 256, 256, 0, stream>>>(x_sites, w_as, al_s, NSITES);
  matvec4<<<(NWELLS + 255) / 256, 256, 0, stream>>>(x_wells, w_ad, al_d, NWELLS);

  // 5. dense GEMMs (bf16 WMMA, f32 accumulate)
  {
    dim3 blk(32, 8);
    dim3 g1((NSITES / 16 + 7) / 8, 256 / 16);  // xs = x_sites @ W_sw
    wmma_gemm_k64<256, false, false><<<g1, blk, 0, stream>>>(ApkSites, BpkSW, nullptr, nullptr,
                                                             xs, NSITES);
    dim3 g2((NWELLS / 16 + 7) / 8, 64 / 16);   // acc_w = x_wells @ (Wl_ww+Wr_ww) + bias
    wmma_gemm_k64<64, false, true><<<g2, blk, 0, stream>>>(ApkWells, BpkWW, nullptr, biasWell,
                                                           acc_w, NWELLS);
    dim3 g3((NSITES / 16 + 7) / 8, 64 / 16);   // site_lin = x_sites @ (Wr_ws+mean W_ss) + bias
    wmma_gemm_k64<64, false, true><<<g3, blk, 0, stream>>>(ApkSites, BpkSite, nullptr, biasSite,
                                                           site_lin, NSITES);
  }

  // 6. GAT edge softmax + scatter into acc_w (already holds wells self term)
  gat_edge_max<<<(NEDGES + 255) / 256, 256, 0, stream>>>(e_sw_src, e_sw_dst, al_s, al_d,
                                                         mmax, NEDGES);
  gat_edge_exp<<<(NEDGES + 255) / 256, 256, 0, stream>>>(e_sw_src, e_sw_dst, al_s, al_d, mmax,
                                                         exbuf, denom, NEDGES);
  recip_scale<<<(NWELLS * 4 + 255) / 256, 256, 0, stream>>>(denom, (long long)NWELLS * 4);
  {
    long long tot = (long long)NEDGES * 64;
    gat_edge_scatter<<<(unsigned)((tot + 255) / 256), 256, 0, stream>>>(
        e_sw_src, e_sw_dst, xs, exbuf, denom, acc_w, tot);
  }

  // 7. SAGE wells -> sites: scatter-sum + count, mean, then += mean @ Wl_ws (WMMA, C-chained)
  {
    long long tot = (long long)NEDGES * 64;
    sage_edge<<<(unsigned)((tot + 255) / 256), 256, 0, stream>>>(
        e_ws_src, e_ws_dst, x_wells, ssum, cnt, tot);
    long long tot2 = (long long)NSITES * 64;
    sage_mean<<<(unsigned)((tot2 + 255) / 256), 256, 0, stream>>>(ssum, cnt, tot2);
    pack_A<<<(NSITES * 8 + 255) / 256, 256, 0, stream>>>(ssum, ApkMean, NSITES);
    dim3 blk(32, 8);
    dim3 g4((NSITES / 16 + 7) / 8, 64 / 16);
    wmma_gemm_k64<64, true, false><<<g4, blk, 0, stream>>>(ApkMean, BpkLWS, site_lin, nullptr,
                                                           site_lin, NSITES);
  }

  // 8. output heads
  head_kernel<<<(NWELLS + 255) / 256, 256, 0, stream>>>(acc_w, Wg, bg, o_gw, NWELLS);
  head_kernel<<<(NSITES + 255) / 256, 256, 0, stream>>>(site_lin, Wsit, bsit, o_si, NSITES);
}
